// SparseConvNet_12532714570270
// MI455X (gfx1250) — matrix operand order repaired and checked by reading, not compiled, not run on hardware
//
#include <hip/hip_runtime.h>

// ---------------------------------------------------------------------------
// CDNA5 (gfx1250) sparse-conv-net as dense implicit-GEMM over bf16 WMMA.
//   - activations channel-last bf16 in HALO-PADDED (D+2)^3 ping-pong buffers
//     -> zero boundary checks; every load is base + compile-time immediate
//   - weights pre-transposed to [28][Cout][Cin] bf16, tap 27 zero-filled
//     -> no K-padding guards either
//   - each wave32 computes an MT x NT grid of 16x16 output tiles via
//     v_wmma_f32_16x16x32_bf16, fully compile-time specialized per layer
//   - BN(scale/shift) + ReLU + mask fused in epilogue
// ---------------------------------------------------------------------------

typedef __bf16 v8bf  __attribute__((ext_vector_type(8)));
typedef __bf16 v16bf __attribute__((ext_vector_type(16)));
typedef float  v8f   __attribute__((ext_vector_type(8)));

constexpr int ilog2c(int v) { return v <= 1 ? 0 : 1 + ilog2c(v / 2); }

__device__ __forceinline__ v8bf zero8() { v8bf z = {}; return z; }

__device__ __forceinline__ v8bf load8(const __bf16* p) {
    return *reinterpret_cast<const v8bf*>(p);  // 16B-aligned by construction
}

// A-fragment 8-wide K-run: all of tap/dz/dy/dx/channel-extra constant-fold
// under the unrolled K loop -> load at aBase + constant immediate offset.
template <int CINL, int PDI, int STRIDE>
__device__ __forceinline__ v8bf a_load(const __bf16* __restrict__ aBase, int k,
                                       int r01, int m) {
    constexpr int CIN = 1 << CINL;
    int tap, cx;
    if (CINL == 4) {            // K chunk = 2 taps
        tap = 2 * k + r01;
        cx  = 0;
    } else if (CINL == 5) {     // K chunk = 1 tap
        tap = k;
        cx  = 16 * r01;
    } else {                    // K chunk = half tap
        tap = k >> 1;
        cx  = 32 * (k & 1) + 16 * r01;
    }
    if (tap >= 27) return zero8();  // compile-time (CINL==4, k==13, r01==1)
    const int dz = tap / 9;
    const int r9 = tap - dz * 9;
    const int dy = r9 / 3;
    const int dx = r9 - dy * 3;
    const int off =
        ((dz * PDI + dy) * PDI + dx) * CIN + cx + m * 16 * STRIDE * CIN;
    return load8(aBase + off);
}

// One wave -> MT x NT tiles of 16(M: output-x) x 16(N: output channel).
template <int CINL, int COUT, int DIN, int DOUT, int STRIDE, int MT, int NT,
          bool FINAL>
__global__ __launch_bounds__(256) void conv_wmma(
    const __bf16* __restrict__ act,    // [(DIN+2)^3][CIN] padded channel-last
    const __bf16* __restrict__ wt,     // [28][COUT][CIN], tap 27 = zeros
    const float*  __restrict__ scale,  // [COUT]
    const float*  __restrict__ shift,  // [COUT]
    const float*  __restrict__ maskv,  // [DOUT^3] (unpadded)
    __bf16*       __restrict__ outAct, // [(DOUT+2)^3][COUT] padded (!FINAL)
    float*        __restrict__ outF) { // NCDHW unpadded       (FINAL)
    constexpr int CIN = 1 << CINL;
    constexpr int PDI = DIN + 2;
    constexpr int PDO = DOUT + 2;
    constexpr int XW  = 16 * MT;
    constexpr int XT  = DOUT / XW;
    constexpr int LXT = ilog2c(XT);
    constexpr int LD  = ilog2c(DOUT);

    const int lane = threadIdx.x & 31;
    const int wid  = blockIdx.x * (blockDim.x >> 5) + (threadIdx.x >> 5);
    const int x0   = (wid & (XT - 1)) * XW;
    const int s    = wid >> LXT;
    const int oy   = s & (DOUT - 1);
    const int oz   = s >> LD;

    const int hl  = lane >> 4;  // wave half
    const int n16 = lane & 15;

    // Per-lane base pointers; everything else is an immediate offset.
    const __bf16* aBase =
        act + (((size_t)(oz * STRIDE) * PDI + oy * STRIDE) * PDI +
               (x0 + n16) * STRIDE) * CIN + 8 * hl;
    const __bf16* bBase =
        wt + (size_t)n16 * CIN + (size_t)8 * hl;
    __builtin_prefetch(bBase, 0, 0);  // global_prefetch_b8

    v8f acc[MT][NT] = {};

    constexpr int TOTK = 27 << CINL;
    constexpr int NK   = (TOTK + 31) >> 5;

#pragma unroll
    for (int k = 0; k < NK; ++k) {
        // ---- A fragments: two 8-wide K-runs per lane (ISA bf16 layout)
        v16bf A[MT];
#pragma unroll
        for (int m = 0; m < MT; ++m) {
            v8bf a0 = a_load<CINL, PDI, STRIDE>(aBase, k, 0, m);
            v8bf a1 = a_load<CINL, PDI, STRIDE>(aBase, k, 1, m);
            A[m] = __builtin_shufflevector(a0, a1, 0, 1, 2, 3, 4, 5, 6, 7, 8, 9,
                                           10, 11, 12, 13, 14, 15);
        }
        // ---- B fragments: two 8-wide K-runs per lane (ISA bf16 layout)
        const int tapb = (CINL == 4) ? 2 * k : (CINL == 5 ? k : (k >> 1));
        const int cxb  = (CINL == 6) ? 32 * (k & 1) : 0;
        v16bf B[NT];
#pragma unroll
        for (int nt = 0; nt < NT; ++nt) {
            const int offB = (tapb * COUT + nt * 16) * CIN + cxb;
            v8bf b0 = load8(bBase + offB);
            v8bf b1 = load8(bBase + offB + (CINL == 4 ? COUT * CIN : 16));
            B[nt] = __builtin_shufflevector(b0, b1, 0, 1, 2, 3, 4, 5, 6, 7, 8,
                                            9, 10, 11, 12, 13, 14, 15);
        }
        // ---- D += A x B  (v_wmma_f32_16x16x32_bf16)
#pragma unroll
        for (int m = 0; m < MT; ++m)
#pragma unroll
            for (int nt = 0; nt < NT; ++nt)
                acc[m][nt] = __builtin_amdgcn_wmma_f32_16x16x32_bf16(
                    false, A[m], false, B[nt], (short)0, acc[m][nt], false,
                    false);
    }

    // ---- Epilogue: BN fold + ReLU + mask. D row M = r + 8*hl, col N = n16.
    float sc[NT], sh[NT];
#pragma unroll
    for (int nt = 0; nt < NT; ++nt) {
        sc[nt] = scale[nt * 16 + n16];
        sh[nt] = shift[nt * 16 + n16];
    }
    const float* mrow = maskv + ((size_t)oz * DOUT + oy) * DOUT + x0 + 8 * hl;
    float*  oBase = FINAL ? outF + (size_t)n16 * (DOUT * DOUT * DOUT) +
                                ((size_t)oz * DOUT + oy) * DOUT + x0 + 8 * hl
                          : nullptr;
    __bf16* sBase = FINAL ? nullptr
                          : outAct + (((size_t)(oz + 1) * PDO + (oy + 1)) * PDO +
                                      (x0 + 1 + 8 * hl)) * COUT + n16;
#pragma unroll
    for (int m = 0; m < MT; ++m) {
#pragma unroll
        for (int r = 0; r < 8; ++r) {
            const float mk = mrow[16 * m + r];
#pragma unroll
            for (int nt = 0; nt < NT; ++nt) {
                const float v = fmaxf(acc[m][nt][r] * sc[nt] + sh[nt], 0.0f) * mk;
                if (FINAL) {
                    oBase[(size_t)nt * 16 * DOUT * DOUT * DOUT + 16 * m + r] = v;
                } else {
                    sBase[(16 * m + r) * COUT + nt * 16] = (__bf16)v;
                }
            }
        }
    }
}

// x (NCDHW f32, zero at inactive sites) -> padded channel-last bf16 (halo=0)
__global__ void prep_x_pad(const float* __restrict__ x,
                           __bf16* __restrict__ act) {
    const int pz = blockIdx.z;
    const int py = blockIdx.y;
    const int t  = blockIdx.x * blockDim.x + threadIdx.x;
    if (t >= 130 * 16) return;
    const int px = t >> 4;
    const int c  = t & 15;
    float v = 0.0f;
    if (px >= 1 && px <= 128 && py >= 1 && py <= 128 && pz >= 1 && pz <= 128)
        v = x[((size_t)c * 128 + (pz - 1)) * 128 * 128 + (py - 1) * 128 +
              (px - 1)];
    act[(((size_t)pz * 130 + py) * 130 + px) * 16 + c] = (__bf16)v;
}

__global__ void prep_mask(const unsigned char* __restrict__ mask,
                          float* __restrict__ m0, int n) {
    const int i = blockIdx.x * blockDim.x + threadIdx.x;
    if (i < n) m0[i] = mask[i] ? 1.0f : 0.0f;
}

// w[3][3][3][Cin][Cout] f32 -> wt[28][Cout][Cin] bf16 (tap 27 zero-filled)
template <int CINL, int COUTL>
__global__ void prep_w(const float* __restrict__ w, __bf16* __restrict__ wt) {
    constexpr int CIN  = 1 << CINL;
    constexpr int COUT = 1 << COUTL;
    const int i = blockIdx.x * blockDim.x + threadIdx.x;
    if (i >= (28 << (CINL + COUTL))) return;
    const int tap = i >> (CINL + COUTL);
    const int rem = i & ((1 << (CINL + COUTL)) - 1);
    const int ci  = rem >> COUTL;
    const int co  = rem & (COUT - 1);
    const float v = (tap < 27) ? w[((size_t)tap * CIN + ci) * COUT + co] : 0.0f;
    wt[((size_t)tap * COUT + co) * CIN + ci] = (__bf16)v;
}

// SparseConv3d(k=3,s=2,p=1) activity: 3^3 max-pool, stride 2, pad 1
template <int DI, int DO>
__global__ void mask_down(const float* __restrict__ mi,
                          float* __restrict__ mo) {
    constexpr int LDO = ilog2c(DO);
    const int i = blockIdx.x * blockDim.x + threadIdx.x;
    if (i >= DO * DO * DO) return;
    const int ox = i & (DO - 1);
    const int t  = i >> LDO;
    const int oy = t & (DO - 1);
    const int oz = t >> LDO;
    float v = 0.0f;
#pragma unroll
    for (int dz = 0; dz < 3; ++dz)
#pragma unroll
        for (int dy = 0; dy < 3; ++dy)
#pragma unroll
            for (int dx = 0; dx < 3; ++dx) {
                const int iz = 2 * oz + dz - 1;
                const int iy = 2 * oy + dy - 1;
                const int ix = 2 * ox + dx - 1;
                if ((unsigned)iz < (unsigned)DI && (unsigned)iy < (unsigned)DI &&
                    (unsigned)ix < (unsigned)DI)
                    v = fmaxf(v, mi[((size_t)iz * DI + iy) * DI + ix]);
            }
    mo[i] = v;
}

extern "C" void kernel_launch(void* const* d_in, const int* in_sizes, int n_in,
                              void* d_out, int out_size, void* d_ws,
                              size_t ws_size, hipStream_t stream) {
    (void)in_sizes; (void)n_in; (void)out_size; (void)ws_size;

    const float*         x    = (const float*)d_in[0];
    const unsigned char* mask = (const unsigned char*)d_in[1];  // jax bool = 1B

    struct L { int wi, si, bi, CinL, CoutL; };
    const L Ls[9] = {
        {2, 3, 4, 4, 4},    {5, 6, 7, 4, 4},    {8, 9, 10, 4, 5},
        {11, 12, 13, 5, 5}, {14, 15, 16, 5, 5}, {17, 18, 19, 5, 6},
        {20, 21, 22, 6, 6}, {23, 24, 25, 6, 6}, {26, 27, 28, 6, 6}};

    // ---- workspace carve-up (256B aligned)
    char*  ws  = (char*)d_ws;
    size_t off = 0;
    auto carve = [&](size_t bytes) {
        void* p = ws + off;
        off = (off + bytes + 255) & ~(size_t)255;
        return p;
    };
    const size_t P0 = (size_t)130 * 130 * 130 * 16 * 2;  // 70.3 MB
    const size_t P1 = (size_t)66 * 66 * 66 * 32 * 2;     // 18.4 MB
    const size_t P2 = (size_t)34 * 34 * 34 * 64 * 2;     //  5.0 MB
    __bf16* actA = (__bf16*)carve(P0);
    __bf16* actB = (__bf16*)carve(P0);
    float*  m0   = (float*)carve((size_t)2097152 * 4);  // 128^3
    float*  m1   = (float*)carve((size_t)262144 * 4);   // 64^3
    float*  m2   = (float*)carve((size_t)32768 * 4);    // 32^3
    __bf16* wt[9];
    for (int l = 0; l < 9; ++l)
        wt[l] = (__bf16*)carve((size_t)28 * 2 << (Ls[l].CinL + Ls[l].CoutL));

    // ---- prep
    prep_x_pad<<<dim3(9, 130, 130), 256, 0, stream>>>(x, actA);
    prep_mask<<<8192, 256, 0, stream>>>(mask, m0, 2097152);
    prep_w<4, 4><<<(28 * 256 + 255) / 256, 256, 0, stream>>>(
        (const float*)d_in[2], wt[0]);
    prep_w<4, 4><<<(28 * 256 + 255) / 256, 256, 0, stream>>>(
        (const float*)d_in[5], wt[1]);
    prep_w<4, 5><<<(28 * 512 + 255) / 256, 256, 0, stream>>>(
        (const float*)d_in[8], wt[2]);
    prep_w<5, 5><<<(28 * 1024 + 255) / 256, 256, 0, stream>>>(
        (const float*)d_in[11], wt[3]);
    prep_w<5, 5><<<(28 * 1024 + 255) / 256, 256, 0, stream>>>(
        (const float*)d_in[14], wt[4]);
    prep_w<5, 6><<<(28 * 2048 + 255) / 256, 256, 0, stream>>>(
        (const float*)d_in[17], wt[5]);
    prep_w<6, 6><<<(28 * 4096 + 255) / 256, 256, 0, stream>>>(
        (const float*)d_in[20], wt[6]);
    prep_w<6, 6><<<(28 * 4096 + 255) / 256, 256, 0, stream>>>(
        (const float*)d_in[23], wt[7]);
    prep_w<6, 6><<<(28 * 4096 + 255) / 256, 256, 0, stream>>>(
        (const float*)d_in[26], wt[8]);
    mask_down<128, 64><<<1024, 256, 0, stream>>>(m0, m1);
    mask_down<64, 32><<<128, 256, 0, stream>>>(m1, m2);

    const float* SC[9];
    const float* SH[9];
    for (int l = 0; l < 9; ++l) {
        SC[l] = (const float*)d_in[Ls[l].si];
        SH[l] = (const float*)d_in[Ls[l].bi];
    }
    float* outF = (float*)d_out;  // [64][32][32][32] NCDHW

    // ---- conv chain; memset zeroes the halo whenever a buffer is about to be
    // rewritten under a new padded-layout interpretation (graph-capturable).
    hipMemsetAsync(actB, 0, P0, stream);
    conv_wmma<4, 16, 128, 128, 1, 4, 1, false><<<4096, 256, 0, stream>>>(
        actA, wt[0], SC[0], SH[0], m0, actB, nullptr);           // subm0 a
    conv_wmma<4, 16, 128, 128, 1, 4, 1, false><<<4096, 256, 0, stream>>>(
        actB, wt[1], SC[1], SH[1], m0, actA, nullptr);           // subm0 b
    hipMemsetAsync(actB, 0, P1, stream);
    conv_wmma<4, 32, 128, 64, 2, 2, 2, false><<<1024, 256, 0, stream>>>(
        actA, wt[2], SC[2], SH[2], m1, actB, nullptr);           // down0
    hipMemsetAsync(actA, 0, P1, stream);
    conv_wmma<5, 32, 64, 64, 1, 2, 2, false><<<1024, 256, 0, stream>>>(
        actB, wt[3], SC[3], SH[3], m1, actA, nullptr);           // subm1 a
    conv_wmma<5, 32, 64, 64, 1, 2, 2, false><<<1024, 256, 0, stream>>>(
        actA, wt[4], SC[4], SH[4], m1, actB, nullptr);           // subm1 b
    hipMemsetAsync(actA, 0, P2, stream);
    conv_wmma<5, 64, 64, 32, 2, 1, 4, false><<<256, 256, 0, stream>>>(
        actB, wt[5], SC[5], SH[5], m2, actA, nullptr);           // down1
    hipMemsetAsync(actB, 0, P2, stream);
    conv_wmma<6, 64, 32, 32, 1, 1, 4, false><<<256, 256, 0, stream>>>(
        actA, wt[6], SC[6], SH[6], m2, actB, nullptr);           // subm2 a
    conv_wmma<6, 64, 32, 32, 1, 1, 4, false><<<256, 256, 0, stream>>>(
        actB, wt[7], SC[7], SH[7], m2, actA, nullptr);           // subm2 b
    conv_wmma<6, 64, 32, 32, 1, 1, 4, true><<<256, 256, 0, stream>>>(
        actA, wt[8], SC[8], SH[8], m2, nullptr, outF);           // subm2 c
}
